// AttentionBlock_81776177316343
// MI455X (gfx1250) — matrix operand order repaired and checked by reading, not compiled, not run on hardware
//
#include <hip/hip_runtime.h>
#include <hip/hip_bf16.h>

#define BATCH 4
#define SEQ   2048
#define HID   2048

typedef __attribute__((ext_vector_type(16))) __bf16 v16bf;
typedef __attribute__((ext_vector_type(8)))  float  v8f;
typedef unsigned int v4u  __attribute__((ext_vector_type(4)));
typedef int          v8i_ __attribute__((ext_vector_type(8)));
typedef int          v4i_ __attribute__((ext_vector_type(4)));

union Frag16 { v16bf v; unsigned int u[8]; };

static __device__ __forceinline__ unsigned short f2bf(float f) {
    unsigned int u = __float_as_uint(f);
    u = (u + 0x7FFFu + ((u >> 16) & 1u)) >> 16;   // round-to-nearest-even
    return (unsigned short)u;
}

// A fragment: lane holds row m=lane%16; VGPR i -> k = 16*(i/4) + 8*(lane/16) + 2*(i%4).
static __device__ __forceinline__ v16bf load_a_frag(const unsigned short* base, int stride,
                                                    int k0, int lane) {
    Frag16 f;
    const unsigned short* p = base + (lane & 15) * stride + k0 + (((lane >> 4) & 1) << 3);
#pragma unroll
    for (int i = 0; i < 8; ++i) {
        int k = ((i >> 2) << 4) + ((i & 3) << 1);
        f.u[i] = *(const unsigned int*)(p + k);
    }
    return f.v;
}

// B fragment from N-major buffer: lane holds col n=lane%16; VGPR j -> k = 16*(j/4) + 8*(lane/16) + 2*(j%4).
static __device__ __forceinline__ v16bf load_b_frag(const unsigned short* base, int stride,
                                                    int k0, int lane) {
    Frag16 f;
    const unsigned short* p = base + (lane & 15) * stride + k0 + (((lane >> 4) & 1) << 3);
#pragma unroll
    for (int j = 0; j < 8; ++j)
        f.u[j] = *(const unsigned int*)(p + ((j >> 2) << 4) + ((j & 3) << 1));
    return f.v;
}

static __device__ __forceinline__ v8f wmma_bf16(v16bf a, v16bf b, v8f c) {
    return __builtin_amdgcn_wmma_f32_16x16x32_bf16(false, a, false, b, (short)0, c, false, false);
}

// ---- Tensor Data Mover: 2D bf16 tile (tile_d0 x tile_d1) global -> LDS ----
// Descriptor layout per CDNA5 ISA ch.8: group0 = {flags, lds_addr, gaddr_lo, gaddr_hi|type},
// group1 = {data_size, tensor dims, tile dims, dim0 stride}.
static __device__ __forceinline__ void tdm_load_2d(unsigned lds_off,
                                                   const unsigned short* gptr,
                                                   unsigned tensor_d0, unsigned tensor_d1,
                                                   unsigned tile_d0, unsigned tile_d1,
                                                   unsigned stride0_elems) {
    unsigned long long ga = (unsigned long long)(size_t)gptr;
    v4u g0;
    g0[0] = 1u;                                                 // count=1, user descriptor
    g0[1] = lds_off;                                            // LDS byte address
    g0[2] = (unsigned)(ga & 0xFFFFFFFFu);                       // global addr [31:0]
    g0[3] = (unsigned)((ga >> 32) & 0x01FFFFFFu) | (2u << 30);  // addr [56:32] | type=2
    v8i_ g1;
    g1[0] = (int)(1u << 16);                                    // data_size=1 (2 bytes)
    g1[1] = (int)((tensor_d0 & 0xFFFFu) << 16);                 // tensor_dim0 [15:0]
    g1[2] = (int)(((tensor_d0 >> 16) & 0xFFFFu) |               // tensor_dim0 [31:16]
                  ((tensor_d1 & 0xFFFFu) << 16));               // tensor_dim1 [15:0]
    g1[3] = (int)(((tensor_d1 >> 16) & 0xFFFFu) |               // tensor_dim1 [31:16]
                  ((tile_d0 & 0xFFFFu) << 16));                 // tile_dim0
    g1[4] = (int)(tile_d1 & 0xFFFFu);                           // tile_dim1 (tile_dim2=0)
    g1[5] = (int)stride0_elems;                                 // tensor_dim0_stride [31:0]
    g1[6] = 0;
    g1[7] = 0;
    v4i_ z4 = {0, 0, 0, 0};
#if defined(__clang_major__) && (__clang_major__ >= 23)
    v8i_ z8 = {0, 0, 0, 0, 0, 0, 0, 0};
    __builtin_amdgcn_tensor_load_to_lds(g0, g1, z4, z4, z8, 0);
#else
    __builtin_amdgcn_tensor_load_to_lds(g0, g1, z4, z4, 0);
#endif
}

static __device__ __forceinline__ unsigned lds_offset_of(const void* p) {
    // generic LDS addresses carry the byte offset in the low 32 bits (aperture in high bits)
    return (unsigned)(size_t)p;
}

// ---------------- Kernel 1: QKV projection -------------------------------
// C[M=B*S, N=3H] = X @ Wqkv^T + b ; Q,K stored [t][h] bf16, V stored [b][h][s] bf16.
__global__ __launch_bounds__(256) void qkv_kernel(const float* __restrict__ X,
                                                  const float* __restrict__ W,
                                                  const float* __restrict__ bias,
                                                  unsigned short* __restrict__ q_ws,
                                                  unsigned short* __restrict__ k_ws,
                                                  unsigned short* __restrict__ vt_ws) {
    __shared__ unsigned short Xs[64][64];   // [token][k] bf16
    __shared__ unsigned short Wsh[64][64];  // [channel][k] bf16 (B^T layout)

    const int tid  = threadIdx.x;
    const int lane = tid & 31;
    const int wave = tid >> 5;
    const int waveM = wave >> 1, waveN = wave & 1;   // 4 x 2 wave grid; wave tile 16x32
    const int row0 = blockIdx.y * 64;
    const int col0 = blockIdx.x * 64;

    v8f acc0 = {0.f, 0.f, 0.f, 0.f, 0.f, 0.f, 0.f, 0.f};
    v8f acc1 = {0.f, 0.f, 0.f, 0.f, 0.f, 0.f, 0.f, 0.f};

    const int sr = tid >> 2, sc = (tid & 3) << 4;    // staging: 16 contiguous floats/thread

    for (int k0 = 0; k0 < HID; k0 += 64) {
        {
            const float* src = X + (size_t)(row0 + sr) * HID + k0 + sc;
#pragma unroll
            for (int v = 0; v < 4; ++v) {
                float4 x = *(const float4*)(src + v * 4);
                Xs[sr][sc + v * 4 + 0] = f2bf(x.x);
                Xs[sr][sc + v * 4 + 1] = f2bf(x.y);
                Xs[sr][sc + v * 4 + 2] = f2bf(x.z);
                Xs[sr][sc + v * 4 + 3] = f2bf(x.w);
            }
        }
        {
            const float* src = W + (size_t)(col0 + sr) * HID + k0 + sc;
#pragma unroll
            for (int v = 0; v < 4; ++v) {
                float4 x = *(const float4*)(src + v * 4);
                Wsh[sr][sc + v * 4 + 0] = f2bf(x.x);
                Wsh[sr][sc + v * 4 + 1] = f2bf(x.y);
                Wsh[sr][sc + v * 4 + 2] = f2bf(x.z);
                Wsh[sr][sc + v * 4 + 3] = f2bf(x.w);
            }
        }
        __syncthreads();
#pragma unroll
        for (int kk = 0; kk < 64; kk += 32) {
            v16bf a  = load_a_frag(&Xs[waveM * 16][0], 64, kk, lane);
            v16bf b0 = load_b_frag(&Wsh[waveN * 32][0], 64, kk, lane);
            v16bf b1 = load_b_frag(&Wsh[waveN * 32 + 16][0], 64, kk, lane);
            acc0 = wmma_bf16(a, b0, acc0);
            acc1 = wmma_bf16(a, b1, acc1);
        }
        __syncthreads();
    }

    const int mhi = ((lane >> 4) & 1) << 3;
#pragma unroll
    for (int t = 0; t < 2; ++t) {
        const v8f acc = t ? acc1 : acc0;
        const int nglob = col0 + waveN * 32 + t * 16 + (lane & 15);
        const float bv = bias[nglob];
        const int region = nglob >> 11;   // 0:Q 1:K 2:V
        const int ch = nglob & (HID - 1);
#pragma unroll
        for (int i = 0; i < 8; ++i) {
            int tok = row0 + waveM * 16 + mhi + i;
            unsigned short val = f2bf(acc[i] + bv);
            if (region == 0)      q_ws[(size_t)tok * HID + ch] = val;
            else if (region == 1) k_ws[(size_t)tok * HID + ch] = val;
            else {
                int bb = tok >> 11, s = tok & (SEQ - 1);
                vt_ws[((size_t)bb * HID + ch) * SEQ + s] = val;
            }
        }
    }
}

// ---------------- Kernel 2: attention ------------------------------------
// One block per (batch, 16-query tile); full fp32 score row kept in LDS.
// Q tile + per-wave K/V panels staged through the Tensor Data Mover.
__global__ __launch_bounds__(256) void attn_kernel(const unsigned short* __restrict__ q_ws,
                                                   const unsigned short* __restrict__ k_ws,
                                                   const unsigned short* __restrict__ vt_ws,
                                                   unsigned short* __restrict__ ctx_ws) {
    __shared__ unsigned short qp_smem[16 * SEQ];          // Q bf16, later P bf16 (64KB)
    __shared__ float          s_smem[16 * SEQ];           // fp32 scores (128KB)
    __shared__ unsigned short kv_smem[8][2][16 * 128];    // per-wave double buffers (64KB)

    const int tid  = threadIdx.x;
    const int lane = tid & 31;
    const int wave = tid >> 5;
    const int b  = blockIdx.x >> 7;
    const int q0 = (blockIdx.x & 127) << 4;
    const int mhi = ((lane >> 4) & 1) << 3;

    // ---- TDM: Q tile (16 x HID bf16) -> LDS, issued by wave 0 only ----
    if (wave == 0) {
        tdm_load_2d(lds_offset_of(qp_smem),
                    q_ws + ((size_t)(b * SEQ + q0)) * HID,
                    HID, BATCH * SEQ, HID, 16, HID);
        __builtin_amdgcn_s_wait_tensorcnt(0);
    }
    __syncthreads();

    const float scale = 0.022097086912079610f;  // 1/sqrt(2048)

    // ---- phase 1: scores = Q @ K^T * scale; wave owns 256 key columns ----
    for (int kt = 0; kt < 16; ++kt) {
        const int n0 = wave * 256 + kt * 16;
        const unsigned short* Kb = k_ws + ((size_t)(b * SEQ + n0)) * HID;
        v8f acc = {0.f, 0.f, 0.f, 0.f, 0.f, 0.f, 0.f, 0.f};

        tdm_load_2d(lds_offset_of(&kv_smem[wave][0][0]), Kb, HID, 16, 128, 16, HID);
        for (int c = 0; c < 16; ++c) {                 // 16 h-chunks of 128
            if (c < 15) {
                tdm_load_2d(lds_offset_of(&kv_smem[wave][(c + 1) & 1][0]),
                            Kb + (c + 1) * 128, HID, 16, 128, 16, HID);
                __builtin_amdgcn_s_wait_tensorcnt(1);  // chunk c has landed
            } else {
                __builtin_amdgcn_s_wait_tensorcnt(0);
            }
            const unsigned short* kbuf = &kv_smem[wave][c & 1][0];
#pragma unroll
            for (int kk = 0; kk < 128; kk += 32) {
                v16bf a  = load_a_frag(qp_smem, HID, c * 128 + kk, lane);
                v16bf bf = load_b_frag(kbuf, 128, kk, lane);
                acc = wmma_bf16(a, bf, acc);
            }
        }
        const int n = n0 + (lane & 15);
#pragma unroll
        for (int i = 0; i < 8; ++i)
            s_smem[(mhi + i) * SEQ + n] = acc[i] * scale;
    }
    __syncthreads();

    // ---- phase 2: softmax per row (16 threads/row, 16-lane shuffle reduce) ----
    {
        const int r = tid >> 4, c0 = tid & 15;
        float* row = &s_smem[r * SEQ];
        float mx = -3.4e38f;
        for (int c = c0; c < SEQ; c += 16) mx = fmaxf(mx, row[c]);
#pragma unroll
        for (int o = 8; o >= 1; o >>= 1) mx = fmaxf(mx, __shfl_xor(mx, o, 16));
        float sum = 0.f;
        for (int c = c0; c < SEQ; c += 16) {
            float e = __expf(row[c] - mx);
            row[c] = e;
            sum += e;
        }
#pragma unroll
        for (int o = 8; o >= 1; o >>= 1) sum += __shfl_xor(sum, o, 16);
        float inv = 1.0f / sum;
        for (int c = c0; c < SEQ; c += 16)
            qp_smem[r * SEQ + c] = f2bf(row[c] * inv);   // P bf16, reusing Q buffer
    }
    __syncthreads();

    // ---- phase 3: context = P @ V; wave owns 256 h columns; V stored [h][s] ----
    const unsigned short* Vbase = vt_ws + (size_t)b * HID * SEQ;
    for (int ht = 0; ht < 16; ++ht) {
        const int h0 = wave * 256 + ht * 16;
        const unsigned short* Vb = Vbase + (size_t)h0 * SEQ;
        v8f acc = {0.f, 0.f, 0.f, 0.f, 0.f, 0.f, 0.f, 0.f};

        tdm_load_2d(lds_offset_of(&kv_smem[wave][0][0]), Vb, SEQ, 16, 128, 16, SEQ);
        for (int c = 0; c < 16; ++c) {                 // 16 key-chunks of 128
            if (c < 15) {
                tdm_load_2d(lds_offset_of(&kv_smem[wave][(c + 1) & 1][0]),
                            Vb + (c + 1) * 128, SEQ, 16, 128, 16, SEQ);
                __builtin_amdgcn_s_wait_tensorcnt(1);
            } else {
                __builtin_amdgcn_s_wait_tensorcnt(0);
            }
            const unsigned short* vbuf = &kv_smem[wave][c & 1][0];
#pragma unroll
            for (int kk = 0; kk < 128; kk += 32) {
                v16bf a  = load_a_frag(qp_smem, SEQ, c * 128 + kk, lane);
                v16bf bf = load_b_frag(vbuf, 128, kk, lane);
                acc = wmma_bf16(a, bf, acc);
            }
        }
        const int h = h0 + (lane & 15);
#pragma unroll
        for (int i = 0; i < 8; ++i)
            ctx_ws[((size_t)(b * SEQ + q0 + mhi + i)) * HID + h] = f2bf(acc[i]);
    }
}

// ---------------- Kernel 3: output projection ----------------------------
__global__ __launch_bounds__(256) void outproj_kernel(const unsigned short* __restrict__ ctx_ws,
                                                      const float* __restrict__ W,
                                                      const float* __restrict__ bias,
                                                      float* __restrict__ out) {
    __shared__ unsigned short As[64][64];
    __shared__ unsigned short Wsh[64][64];

    const int tid  = threadIdx.x;
    const int lane = tid & 31;
    const int wave = tid >> 5;
    const int waveM = wave >> 1, waveN = wave & 1;
    const int row0 = blockIdx.y * 64;
    const int col0 = blockIdx.x * 64;

    v8f acc0 = {0.f, 0.f, 0.f, 0.f, 0.f, 0.f, 0.f, 0.f};
    v8f acc1 = {0.f, 0.f, 0.f, 0.f, 0.f, 0.f, 0.f, 0.f};

    const int sr = tid >> 2, sc = (tid & 3) << 4;

    for (int k0 = 0; k0 < HID; k0 += 64) {
        {   // A already bf16: straight 32-byte copies
            const unsigned short* src = ctx_ws + (size_t)(row0 + sr) * HID + k0 + sc;
            *(uint4*)&As[sr][sc]     = *(const uint4*)(src);
            *(uint4*)&As[sr][sc + 8] = *(const uint4*)(src + 8);
        }
        {   // W fp32 -> bf16
            const float* src = W + (size_t)(col0 + sr) * HID + k0 + sc;
#pragma unroll
            for (int v = 0; v < 4; ++v) {
                float4 x = *(const float4*)(src + v * 4);
                Wsh[sr][sc + v * 4 + 0] = f2bf(x.x);
                Wsh[sr][sc + v * 4 + 1] = f2bf(x.y);
                Wsh[sr][sc + v * 4 + 2] = f2bf(x.z);
                Wsh[sr][sc + v * 4 + 3] = f2bf(x.w);
            }
        }
        __syncthreads();
#pragma unroll
        for (int kk = 0; kk < 64; kk += 32) {
            v16bf a  = load_a_frag(&As[waveM * 16][0], 64, kk, lane);
            v16bf b0 = load_b_frag(&Wsh[waveN * 32][0], 64, kk, lane);
            v16bf b1 = load_b_frag(&Wsh[waveN * 32 + 16][0], 64, kk, lane);
            acc0 = wmma_bf16(a, b0, acc0);
            acc1 = wmma_bf16(a, b1, acc1);
        }
        __syncthreads();
    }

    const int mhi = ((lane >> 4) & 1) << 3;
#pragma unroll
    for (int t = 0; t < 2; ++t) {
        const v8f acc = t ? acc1 : acc0;
        const int n = col0 + waveN * 32 + t * 16 + (lane & 15);
        const float bv = bias[n];
#pragma unroll
        for (int i = 0; i < 8; ++i) {
            int tok = row0 + waveM * 16 + mhi + i;
            out[(size_t)tok * HID + n] = acc[i] + bv;
        }
    }
}

extern "C" void kernel_launch(void* const* d_in, const int* in_sizes, int n_in,
                              void* d_out, int out_size, void* d_ws, size_t ws_size,
                              hipStream_t stream) {
    const float* X    = (const float*)d_in[0];
    const float* wqkv = (const float*)d_in[1];
    const float* bqkv = (const float*)d_in[2];
    const float* wout = (const float*)d_in[3];
    const float* bout = (const float*)d_in[4];
    float* out = (float*)d_out;

    const size_t n = (size_t)BATCH * SEQ * HID;
    unsigned short* q_ws   = (unsigned short*)d_ws;   // 4 bf16 buffers ~134 MB
    unsigned short* k_ws   = q_ws + n;
    unsigned short* vt_ws  = k_ws + n;
    unsigned short* ctx_ws = vt_ws + n;

    const int M = BATCH * SEQ;
    dim3 blk(256);

    qkv_kernel<<<dim3((3 * HID) / 64, M / 64), blk, 0, stream>>>(X, wqkv, bqkv, q_ws, k_ws, vt_ws);
    attn_kernel<<<dim3(BATCH * (SEQ / 16)), blk, 0, stream>>>(q_ws, k_ws, vt_ws, ctx_ws);
    outproj_kernel<<<dim3(HID / 64, M / 64), blk, 0, stream>>>(ctx_ws, wout, bout, out);
}
